// NeuralODE_84164179132487
// MI455X (gfx1250) — compile-verified, matrix operand order checked by reading, not run on hardware
//
#include <hip/hip_runtime.h>
#include <hip/hip_bf16.h>
#include <math.h>

typedef __bf16 bhalf;
typedef __attribute__((ext_vector_type(16))) __bf16 v16bf;
typedef __attribute__((ext_vector_type(8)))  float  v8f;
typedef __attribute__((ext_vector_type(4)))  int    v4i;

typedef __attribute__((address_space(1))) v4i* as1_v4i_ptr;
typedef __attribute__((address_space(3))) v4i* as3_v4i_ptr;

#define TM 128
#define TN 128
#define TK 32
#define LDSS 40                 // LDS row stride in bf16 elems (80B -> rows 16B aligned)
#define BUFE (128 * LDSS)       // elems per LDS tile buffer

#if __has_builtin(__builtin_amdgcn_global_load_async_to_lds_b128)
#define ODE_ASYNC 1
#else
#define ODE_ASYNC 0
#endif

union Frag16 { uint4 u[2]; v16bf v; };

// 16-byte global -> LDS copy, async (ASYNCcnt path) when available.
__device__ __forceinline__ void cp16(const bhalf* g, bhalf* l) {
#if ODE_ASYNC
  __builtin_amdgcn_global_load_async_to_lds_b128(
      (as1_v4i_ptr)g, (as3_v4i_ptr)l, 0, 0);
#else
  *(uint4*)l = *(const uint4*)g;
#endif
}

__device__ __forceinline__ void wait_async() {
#if ODE_ASYNC
#if __has_builtin(__builtin_amdgcn_s_wait_asynccnt)
  __builtin_amdgcn_s_wait_asynccnt(0);
#else
  asm volatile("s_wait_asynccnt 0" ::: "memory");
#endif
#endif
}

// ---------------------------------------------------------------------------
// Unified bf16 WMMA GEMM:  C(MxN) = A(MxK) @ WT(NxK)^T  (+ per-column init)
//   mode 0    : out = tanh(C + bias[n] (+ t*trow[n]))   -> bf16 outb
//   mode 1..4 : k = C + bias[n]; fused RK4 stage epilogue
// 128 threads = 4 wave32s, block tile 128x128, wave tile 64x64,
// double-buffered async global->LDS staging, 16 WMMAs per wave per K-step.
// ---------------------------------------------------------------------------
__global__ __launch_bounds__(128)
void ode_wmma_gemm(const bhalf* __restrict__ A, const bhalf* __restrict__ WT,
                   int M, int N, int K,
                   const float* __restrict__ bias, const float* __restrict__ trow,
                   float tval, int mode,
                   bhalf* __restrict__ outb, float* __restrict__ hbuf,
                   float* __restrict__ ksum, float cnext, float kw, float dt6)
{
  __shared__ alignas(16) bhalf lA[2 * BUFE];
  __shared__ alignas(16) bhalf lB[2 * BUFE];

  const int tid   = threadIdx.x;
  const int lane  = tid & 31;
  const int wid   = tid >> 5;          // 0..3
  const int mwave = (wid >> 1) * 64;   // 2x2 wave grid
  const int nwave = (wid & 1) * 64;
  const int mbase = blockIdx.y * TM;
  const int nbase = blockIdx.x * TN;

  v8f acc[4][4] = {};

  // ---- stage one K-block (TMxTK of A, TNxTK of WT) into LDS buffer ib ----
  auto stage = [&](int kb, int ib) {
    bhalf* la = lA + ib * BUFE;
    bhalf* lb = lB + ib * BUFE;
#pragma unroll
    for (int i = 0; i < 4; ++i) {
      const int s  = tid + 128 * i;    // 512 16-byte slots per tile
      const int r  = s >> 2;
      const int c8 = (s & 3) * 8;
      cp16(A  + (size_t)(mbase + r) * K + kb + c8, la + r * LDSS + c8);
      cp16(WT + (size_t)(nbase + r) * K + kb + c8, lb + r * LDSS + c8);
    }
  };

  stage(0, 0);
  int ib = 0;
  for (int kb = 0; kb < K; kb += TK, ib ^= 1) {
    wait_async();
    __syncthreads();                       // buffer ib visible to all waves
    if (kb + TK < K) stage(kb + TK, ib ^ 1);   // prefetch next K-block

    // ---- fragments per ISA 16-bit layout (lane>>4 selects 8-wide K runs)
    const int lm = lane & 15;
    const int kh = (lane >> 4) * 8;
    const bhalf* la = lA + ib * BUFE;
    const bhalf* lb = lB + ib * BUFE;
    Frag16 af[4], bfr[4];
#pragma unroll
    for (int t = 0; t < 4; ++t) {
      const bhalf* pa = la + (mwave + t * 16 + lm) * LDSS + kh;
      af[t].u[0] = *(const uint4*)pa;
      af[t].u[1] = *(const uint4*)(pa + 16);
      const bhalf* pb = lb + (nwave + t * 16 + lm) * LDSS + kh;
      bfr[t].u[0] = *(const uint4*)pb;
      bfr[t].u[1] = *(const uint4*)(pb + 16);
    }

#pragma unroll
    for (int mt = 0; mt < 4; ++mt)
#pragma unroll
      for (int nt = 0; nt < 4; ++nt)
        acc[mt][nt] = __builtin_amdgcn_wmma_f32_16x16x32_bf16(
            false, af[mt].v, false, bfr[nt].v, (short)0, acc[mt][nt],
            false, false);

    __syncthreads();                       // all waves done reading buffer ib
  }

  // ---- fused epilogue (C/D layout: lane%16 -> N, vgpr r + 8*(lane>>4) -> M)
  const int lm = lane & 15;
  const int mh = (lane >> 4) * 8;
#pragma unroll
  for (int nt = 0; nt < 4; ++nt) {
    const int n = nbase + nwave + nt * 16 + lm;
    float cinit = bias[n];
    if (trow) cinit += tval * trow[n];
#pragma unroll
    for (int mt = 0; mt < 4; ++mt) {
#pragma unroll
      for (int r = 0; r < 8; ++r) {
        const int m = mbase + mwave + mt * 16 + mh + r;
        const size_t idx = (size_t)m * N + n;
        const float v = acc[mt][nt][r] + cinit;
        if (mode == 0) {
          outb[idx] = (bhalf)tanhf(v);                       // hidden layers
        } else if (mode == 4) {                              // final RK stage
          const float ks = ksum[idx] + v;                    // k1+2k2+2k3+k4
          const float hn = hbuf[idx] + dt6 * ks;
          hbuf[idx] = hn;                                    // d_out updated
          outb[idx] = (bhalf)hn;                             // next-step input
        } else {                                             // stages 1..3
          const float ks = (mode == 1) ? v : (ksum[idx] + kw * v);
          ksum[idx] = ks;
          outb[idx] = (bhalf)(hbuf[idx] + cnext * v);        // next-stage input
        }
      }
    }
  }
}

// WT[n*K + k] = bf16(W[k*N + n])   (coalesced writes; runs once per launch)
__global__ void ode_transpose_bf16(const float* __restrict__ W,
                                   bhalf* __restrict__ WT, int K, int N)
{
  const size_t total = (size_t)K * N;
  for (size_t i = (size_t)blockIdx.x * blockDim.x + threadIdx.x; i < total;
       i += (size_t)gridDim.x * blockDim.x) {
    const size_t n = i / K, k = i % K;
    WT[i] = (bhalf)W[k * (size_t)N + n];
  }
}

__global__ void ode_prep(const float* __restrict__ h0, float* __restrict__ hbuf,
                         bhalf* __restrict__ ab, int total)
{
  for (int i = blockIdx.x * blockDim.x + threadIdx.x; i < total;
       i += gridDim.x * blockDim.x) {
    const float v = h0[i];
    hbuf[i] = v;
    ab[i]   = (bhalf)v;
  }
}

extern "C" void kernel_launch(void* const* d_in, const int* in_sizes, int n_in,
                              void* d_out, int out_size, void* d_ws, size_t ws_size,
                              hipStream_t stream)
{
  const float* h0 = (const float*)d_in[0];
  const float* W1 = (const float*)d_in[1];   // (1025, 2048)
  const float* b1 = (const float*)d_in[2];
  const float* W2 = (const float*)d_in[3];   // (2048, 2048)
  const float* b2 = (const float*)d_in[4];
  const float* W3 = (const float*)d_in[5];   // (2048, 1024)
  const float* b3 = (const float*)d_in[6];

  constexpr int B = 4096, H = 1024;
  float* hbuf = (float*)d_out;               // state lives in d_out (f32)

  char* p = (char*)d_ws;
  bhalf* w1t  = (bhalf*)p; p += (size_t)2048 * 1024 * 2;   // WT1 [2048][1024]
  bhalf* w2t  = (bhalf*)p; p += (size_t)2048 * 2048 * 2;   // WT2 [2048][2048]
  bhalf* w3t  = (bhalf*)p; p += (size_t)1024 * 2048 * 2;   // WT3 [1024][2048]
  bhalf* abuf = (bhalf*)p; p += (size_t)B * H * 2;         // current eval input
  bhalf* x1b  = (bhalf*)p; p += (size_t)B * 2 * H * 2;     // layer-1 out (bf16)
  bhalf* x2b  = (bhalf*)p; p += (size_t)B * 2 * H * 2;     // layer-2 out (bf16)
  float* ksum = (float*)p;                                 // RK4 accumulator

  // once-per-launch weight prep + state init (deterministic; re-done every call)
  ode_transpose_bf16<<<1024, 256, 0, stream>>>(W1, w1t, H, 2 * H);
  ode_transpose_bf16<<<1024, 256, 0, stream>>>(W2, w2t, 2 * H, 2 * H);
  ode_transpose_bf16<<<1024, 256, 0, stream>>>(W3, w3t, 2 * H, H);
  ode_prep<<<512, 256, 0, stream>>>(h0, hbuf, abuf, B * H);

  const float dt = 0.1f;
  const float* w1row = W1 + (size_t)H * (2 * H);   // time row of W1 (f32)
  const dim3 blk(128);                             // 4 wave32s
  const dim3 g12(2 * H / TN, B / TM);   // 16 x 32 blocks
  const dim3 g3(H / TN, B / TM);        //  8 x 32 blocks

  for (int i = 0; i < 10; ++i) {
    const float ti = i * dt;
    const struct { float t; int mode; float cn; float kw; } st[4] = {
      { ti,             1, 0.5f * dt, 1.f },
      { ti + 0.5f * dt, 2, 0.5f * dt, 2.f },
      { ti + 0.5f * dt, 3, dt,        2.f },
      { ti + dt,        4, 0.f,       1.f },
    };
    for (int s = 0; s < 4; ++s) {
      // L1: tanh([h|t] @ W1 + b1)  via K=1024 GEMM + (b1 + t*W1row) init
      ode_wmma_gemm<<<g12, blk, 0, stream>>>(abuf, w1t, B, 2 * H, H,
          b1, w1row, st[s].t, 0, x1b, nullptr, nullptr, 0.f, 0.f, 0.f);
      // L2: tanh(x1 @ W2 + b2)
      ode_wmma_gemm<<<g12, blk, 0, stream>>>(x1b, w2t, B, 2 * H, 2 * H,
          b2, nullptr, 0.f, 0, x2b, nullptr, nullptr, 0.f, 0.f, 0.f);
      // L3: k = x2 @ W3 + b3, fused RK4 stage update
      ode_wmma_gemm<<<g3, blk, 0, stream>>>(x2b, w3t, B, H, 2 * H,
          b3, nullptr, 0.f, st[s].mode, abuf, hbuf, ksum,
          st[s].cn, st[s].kw, dt / 6.f);
    }
  }
}
